// MultiHeadAttentionLayer_18906446037363
// MI455X (gfx1250) — matrix operand order repaired and checked by reading, not compiled, not run on hardware
//
#include <hip/hip_runtime.h>
#include <hip/hip_bf16.h>

// MHA: B=4, S=2048, D=1024, H=16, DK=64
// d_out = [out fp32 (B*S*D)] ++ [sim_norm fp32 (B*H*S*S)]

#define BATCH 4
#define SLEN  2048
#define DMODEL 1024
#define NHEAD 16
#define DKH   64

typedef __attribute__((ext_vector_type(16))) __bf16 v16bf;
typedef __attribute__((ext_vector_type(8)))  float  v8f;
typedef __attribute__((vector_size(16)))     int    vec4i;

// gfx1250 async global->LDS path (ASYNCcnt-tracked); guarded so both toolchains compile.
#if defined(__has_builtin)
#if __has_builtin(__builtin_amdgcn_global_load_async_to_lds_b128) && \
    __has_builtin(__builtin_amdgcn_s_wait_asynccnt)
#define USE_ASYNC_LDS 1
#endif
#endif
#ifndef USE_ASYNC_LDS
#define USE_ASYNC_LDS 0
#endif

// Builtin expects pointers to 16-byte int vectors in AS1 (global) / AS3 (LDS).
#define AS1CAST(p) ((__attribute__((address_space(1))) vec4i*)(p))
#define AS3CAST(p) ((__attribute__((address_space(3))) vec4i*)(p))

// ---------------------------------------------------------------------------
// WMMA fragment loaders (wave32, 16x16x32 bf16 shapes; same contiguous k order in A and B)
// A (16x32, MxK) row-major source, row stride `ld` elements:
//   lane: row = lane&15 ; 16 consecutive k-elements starting at (lane>>4)*16
// B (32x16, KxN) from an N-major (transposed) source, row stride `ld`:
//   lane: col = lane&15 ; 16 consecutive k-elements starting at (lane>>4)*16
// ---------------------------------------------------------------------------
__device__ __forceinline__ v16bf load_frag_a(const __bf16* base, int ld, int lane) {
  const int row = lane & 15;
  const int hk  = (lane >> 4) << 4;           // 0 or 16
  const __bf16* p = base + (size_t)row * ld;
  v16bf a;
  ((uint4*)&a)[0] = *(const uint4*)(p + hk);        // k = hk .. hk+7
  ((uint4*)&a)[1] = *(const uint4*)(p + hk + 8);    // k = hk+8 .. hk+15
  return a;
}

__device__ __forceinline__ v16bf load_frag_b(const __bf16* base, int ld, int lane) {
  const int col = lane & 15;
  const int kb  = (lane >> 4) << 4;           // 0 or 16
  const __bf16* p = base + (size_t)col * ld + kb;
  v16bf b;
  ((uint4*)&b)[0] = ((const uint4*)p)[0];
  ((uint4*)&b)[1] = ((const uint4*)p)[1];
  return b;
}

__device__ __forceinline__ v8f wmma_bf16(v16bf a, v16bf b, v8f c) {
  return __builtin_amdgcn_wmma_f32_16x16x32_bf16(false, a, false, b, (short)0, c,
                                                 false, false);
}

// ---------------------------------------------------------------------------
// fp32 -> bf16 elementwise
// ---------------------------------------------------------------------------
__global__ void cvt_bf16_kernel(const float* __restrict__ src,
                                __bf16* __restrict__ dst, int n) {
  int i = blockIdx.x * blockDim.x + threadIdx.x;
  if (i < n) dst[i] = (__bf16)src[i];
}

// fp32 [K,N] -> bf16 transposed [N,K]
__global__ void cvt_bf16_t_kernel(const float* __restrict__ src,
                                  __bf16* __restrict__ dst, int K, int N) {
  int i = blockIdx.x * blockDim.x + threadIdx.x;
  if (i < K * N) {
    int n = i / K, k = i - n * K;
    dst[i] = (__bf16)src[(size_t)k * N + n];
  }
}

// ---------------------------------------------------------------------------
// Stage one 64(n) x 32(k) bf16 tile of Bt (row stride ld) into LDS (row
// stride 32). 128 threads x 2 chunks x 16B = 4KB. Async (ASYNCcnt) when
// available, else direct global->LDS copy.
// ---------------------------------------------------------------------------
__device__ __forceinline__ void stage_b_tile(const __bf16* __restrict__ src, int ld,
                                             __bf16* dst_lds, int tid) {
#pragma unroll
  for (int c = 0; c < 2; ++c) {
    const int chunk = tid + c * 128;        // 0..255 ; 4 chunks per row
    const int row = chunk >> 2;
    const int ko  = (chunk & 3) * 8;
    const __bf16* g = src + (size_t)row * ld + ko;
    __bf16* l = dst_lds + row * 32 + ko;
#if USE_ASYNC_LDS
    __builtin_amdgcn_global_load_async_to_lds_b128(AS1CAST(g), AS3CAST(l), 0, 0);
#else
    *(uint4*)l = *(const uint4*)g;
#endif
  }
}

// ---------------------------------------------------------------------------
// WMMA GEMM: C[M,N] = A[M,K](bf16) * Bt[N,K](bf16, N-major) + bias[N]
// block = 128 threads (4 waves); block tile = 64(M) x 64(N); wave = 16 x 64
// Bt tile double-buffered through LDS (async DMA overlapped with WMMA).
// MODE 0: bf16 out, [B,H,S,DK]   (Q, K projections; m=b*S+s, n=h*DK+dk)
// MODE 1: bf16 out, [B,H,DK,S]   (V projection, pre-transposed)
// MODE 2: fp32 out, [M,N]        (final output projection -> d_out)
// ---------------------------------------------------------------------------
template <int MODE>
__global__ void gemm_wmma_kernel(const __bf16* __restrict__ A,
                                 const __bf16* __restrict__ Bt,
                                 const float* __restrict__ bias,
                                 void* __restrict__ Cout,
                                 int M, int N, int K) {
  __shared__ __bf16 Bbuf[2][64 * 32];   // 2 x 4KB double buffer

  const int tid  = threadIdx.x;
  const int lane = tid & 31;
  const int wave = tid >> 5;
  const int m0 = blockIdx.y * 64 + wave * 16;
  const int n0 = blockIdx.x * 64;

  const __bf16* Arow = A + (size_t)m0 * K;
  const __bf16* Btil = Bt + (size_t)n0 * K;

  const int KT = K / 32;
  stage_b_tile(Btil, K, Bbuf[0], tid);    // prologue: tile 0 in flight

  v8f c[4] = {};
  for (int kt = 0; kt < KT; ++kt) {
    const int kk = kt * 32;
#if USE_ASYNC_LDS
    __builtin_amdgcn_s_wait_asynccnt(0);  // own async stores to LDS done
#endif
    __syncthreads();                      // tile kt visible; prev reads done
    if (kt + 1 < KT)                      // kick next tile; overlaps WMMAs
      stage_b_tile(Btil + (kk + 32), K, Bbuf[(kt + 1) & 1], tid);

    __builtin_prefetch(Arow + kk + 64, 0, 3);   // global_prefetch_b8 on A stream

    v16bf a = load_frag_a(Arow + kk, K, lane);
    const __bf16* bl = Bbuf[kt & 1];
#pragma unroll
    for (int j = 0; j < 4; ++j) {
      v16bf b = load_frag_b(bl + (j * 16) * 32, 32, lane);   // ds reads
      c[j] = wmma_bf16(a, b, c[j]);
    }
  }

  // C layout: lane&15 = column, rows = r + ((lane>>4)<<3)
  const int mb = m0 + ((lane >> 4) << 3);
#pragma unroll
  for (int j = 0; j < 4; ++j) {
    const int n  = n0 + j * 16 + (lane & 15);
    const float bv = bias[n];
#pragma unroll
    for (int r = 0; r < 8; ++r) {
      const int m = mb + r;
      const float v = c[j][r] + bv;
      if (MODE == 0) {
        int b_ = m >> 11, s = m & (SLEN - 1);
        int h  = n >> 6,  dk = n & (DKH - 1);
        ((__bf16*)Cout)[(((size_t)(b_ * NHEAD + h)) * SLEN + s) * DKH + dk] = (__bf16)v;
      } else if (MODE == 1) {
        int b_ = m >> 11, s = m & (SLEN - 1);
        int h  = n >> 6,  dk = n & (DKH - 1);
        ((__bf16*)Cout)[(((size_t)(b_ * NHEAD + h)) * DKH + dk) * SLEN + s] = (__bf16)v;
      } else {
        ((float*)Cout)[(size_t)m * N + n] = v;
      }
    }
  }
}

// ---------------------------------------------------------------------------
// Attention core: one workgroup per (b, h, 16-row q tile).
// 128 threads = 4 waves. Dynamic LDS:
//   sc[16][2048] fp32 scores (128KB) | pb[16][2048] bf16 probs (64KB) | red (1KB)
// Phase 1: waves strip-mine 128 key tiles with WMMA -> scores to LDS (+mask/scale)
// Phase 2: cooperative row softmax; write sim_norm fp32; bf16 probs to LDS
// Phase 3: each wave owns a 16-wide DV slice; WMMA P(LDS) x Vt(global)
// ---------------------------------------------------------------------------
__global__ void attn_kernel(const __bf16* __restrict__ Qbf,
                            const __bf16* __restrict__ Kbf,
                            const __bf16* __restrict__ Vt,
                            const int* __restrict__ mask,
                            float* __restrict__ sim_out,
                            __bf16* __restrict__ xbf) {
  extern __shared__ char smem[];
  float*  sc     = (float*)smem;                              // 16 x 2048 fp32
  __bf16* pb     = (__bf16*)(smem + (size_t)16 * SLEN * 4);   // 16 x 2048 bf16
  float*  redmax = (float*)(smem + (size_t)16 * SLEN * 6);    // 16 x 8
  float*  redsum = redmax + 16 * 8;                           // 16 x 8

  const int wg = blockIdx.x;
  const int qt = wg & 127;
  const int h  = (wg >> 7) & (NHEAD - 1);
  const int b  = wg >> 11;
  const int q0 = qt * 16;

  const int tid  = threadIdx.x;
  const int lane = tid & 31;
  const int wave = tid >> 5;   // 0..3

  const size_t hb = (size_t)(b * NHEAD + h) * SLEN * DKH;  // head base (Q,K,Vt)

  // Q A-fragments for this q tile (DK=64 -> two k-chunks of 32), kept in VGPRs
  const v16bf qa0 = load_frag_a(Qbf + hb + (size_t)q0 * DKH + 0,  DKH, lane);
  const v16bf qa1 = load_frag_a(Qbf + hb + (size_t)q0 * DKH + 32, DKH, lane);

  const int* mrow = mask + b * SLEN;

  // ---- Phase 1: scores ----
  for (int kt = wave; kt < SLEN / 16; kt += 4) {
    const int n0 = kt * 16;
    v16bf kb0 = load_frag_b(Kbf + hb + (size_t)n0 * DKH + 0,  DKH, lane);
    v16bf kb1 = load_frag_b(Kbf + hb + (size_t)n0 * DKH + 32, DKH, lane);
    v8f c = {};
    c = wmma_bf16(qa0, kb0, c);
    c = wmma_bf16(qa1, kb1, c);

    const int col = n0 + (lane & 15);
    const bool masked = (mrow[col] == 0);
    const int rbase = (lane >> 4) << 3;
#pragma unroll
    for (int r = 0; r < 8; ++r) {
      float v = masked ? -1e10f : c[r] * 0.125f;  // 1/sqrt(64)
      sc[(size_t)(rbase + r) * SLEN + col] = v;
    }
  }
  __syncthreads();

  // ---- Phase 2: softmax over each of 16 rows (8 threads x 256 cols per row) ----
  const int row = tid >> 3;
  const int seg = tid & 7;
  float* rp = sc + (size_t)row * SLEN + seg * 256;

  float lmax = -3.4e38f;
  for (int i = 0; i < 256; ++i) lmax = fmaxf(lmax, rp[i]);
  redmax[row * 8 + seg] = lmax;
  __syncthreads();
  float rmax = redmax[row * 8];
#pragma unroll
  for (int i = 1; i < 8; ++i) rmax = fmaxf(rmax, redmax[row * 8 + i]);

  float lsum = 0.f;
  for (int i = 0; i < 256; ++i) {
    float e = __expf(rp[i] - rmax);
    rp[i] = e;
    lsum += e;
  }
  redsum[row * 8 + seg] = lsum;
  __syncthreads();
  float rsum = 0.f;
#pragma unroll
  for (int i = 0; i < 8; ++i) rsum += redsum[row * 8 + i];
  const float inv = 1.0f / rsum;

  const size_t simbase =
      ((size_t)(b * NHEAD + h) * SLEN + q0 + row) * SLEN + seg * 256;
  __bf16* pp = pb + (size_t)row * SLEN + seg * 256;
  for (int i = 0; i < 256; i += 4) {
    float4 p;
    p.x = rp[i + 0] * inv; p.y = rp[i + 1] * inv;
    p.z = rp[i + 2] * inv; p.w = rp[i + 3] * inv;
    *(float4*)(sim_out + simbase + i) = p;
    pp[i + 0] = (__bf16)p.x; pp[i + 1] = (__bf16)p.y;
    pp[i + 2] = (__bf16)p.z; pp[i + 3] = (__bf16)p.w;
  }
  __syncthreads();

  // ---- Phase 3: x = P @ V ; wave owns dv columns [wave*16, wave*16+16) ----
  const int dv0 = wave * 16;
  v8f acc = {};
  for (int kt = 0; kt < SLEN / 32; ++kt) {
    v16bf pa = load_frag_a(pb + kt * 32, SLEN, lane);                       // LDS
    v16bf vb = load_frag_b(Vt + hb + (size_t)dv0 * SLEN + kt * 32, SLEN, lane);
    acc = wmma_bf16(pa, vb, acc);
  }

  const int rbase = (lane >> 4) << 3;
  const int dv = dv0 + (lane & 15);
#pragma unroll
  for (int r = 0; r < 8; ++r) {
    const int s = q0 + rbase + r;
    xbf[((size_t)(b * SLEN + s)) * DMODEL + h * DKH + dv] = (__bf16)acc[r];
  }
}

// ---------------------------------------------------------------------------
// Launch: conversions -> 3 projection GEMMs -> attention -> output GEMM
// Workspace layout (bytes), total ~120 MB:
//   qin/kin/vin bf16 (3 x 16MB) | wqt/wkt/wvt/wot bf16 (4 x 2MB)
//   Qb/Kb/Vt/Xb bf16 (4 x 16MB)
// ---------------------------------------------------------------------------
extern "C" void kernel_launch(void* const* d_in, const int* in_sizes, int n_in,
                              void* d_out, int out_size, void* d_ws, size_t ws_size,
                              hipStream_t stream) {
  (void)in_sizes; (void)n_in; (void)out_size; (void)ws_size;

  const float* query = (const float*)d_in[0];
  const float* key   = (const float*)d_in[1];
  const float* value = (const float*)d_in[2];
  const int*   mask  = (const int*)d_in[3];
  const float* w_q = (const float*)d_in[4];
  const float* b_q = (const float*)d_in[5];
  const float* w_k = (const float*)d_in[6];
  const float* b_k = (const float*)d_in[7];
  const float* w_v = (const float*)d_in[8];
  const float* b_v = (const float*)d_in[9];
  const float* w_o = (const float*)d_in[10];
  const float* b_o = (const float*)d_in[11];

  const size_t ACT = (size_t)BATCH * SLEN * DMODEL;   // 8,388,608
  const size_t WSZ = (size_t)DMODEL * DMODEL;         // 1,048,576

  char* w = (char*)d_ws;
  __bf16* qin = (__bf16*)w;                 w += ACT * 2;
  __bf16* kin = (__bf16*)w;                 w += ACT * 2;
  __bf16* vin = (__bf16*)w;                 w += ACT * 2;
  __bf16* wqt = (__bf16*)w;                 w += WSZ * 2;
  __bf16* wkt = (__bf16*)w;                 w += WSZ * 2;
  __bf16* wvt = (__bf16*)w;                 w += WSZ * 2;
  __bf16* wot = (__bf16*)w;                 w += WSZ * 2;
  __bf16* Qb  = (__bf16*)w;                 w += ACT * 2;
  __bf16* Kb  = (__bf16*)w;                 w += ACT * 2;
  __bf16* Vt  = (__bf16*)w;                 w += ACT * 2;
  __bf16* Xb  = (__bf16*)w;                 w += ACT * 2;

  float* out_proj = (float*)d_out;                    // [B,S,D]
  float* sim_out  = (float*)d_out + ACT;              // [B,H,S,S]

  // 1) conversions
  {
    int n = (int)ACT;
    int blocks = (n + 255) / 256;
    hipLaunchKernelGGL(cvt_bf16_kernel, dim3(blocks), dim3(256), 0, stream, query, qin, n);
    hipLaunchKernelGGL(cvt_bf16_kernel, dim3(blocks), dim3(256), 0, stream, key,   kin, n);
    hipLaunchKernelGGL(cvt_bf16_kernel, dim3(blocks), dim3(256), 0, stream, value, vin, n);
    int nw = (int)WSZ;
    int wblocks = (nw + 255) / 256;
    hipLaunchKernelGGL(cvt_bf16_t_kernel, dim3(wblocks), dim3(256), 0, stream, w_q, wqt, DMODEL, DMODEL);
    hipLaunchKernelGGL(cvt_bf16_t_kernel, dim3(wblocks), dim3(256), 0, stream, w_k, wkt, DMODEL, DMODEL);
    hipLaunchKernelGGL(cvt_bf16_t_kernel, dim3(wblocks), dim3(256), 0, stream, w_v, wvt, DMODEL, DMODEL);
    hipLaunchKernelGGL(cvt_bf16_t_kernel, dim3(wblocks), dim3(256), 0, stream, w_o, wot, DMODEL, DMODEL);
  }

  // 2) projection GEMMs: M=8192, N=1024, K=1024; grid (N/64, M/64)
  {
    dim3 grid(DMODEL / 64, (BATCH * SLEN) / 64);
    dim3 block(128);
    hipLaunchKernelGGL((gemm_wmma_kernel<0>), grid, block, 0, stream,
                       qin, wqt, b_q, (void*)Qb, BATCH * SLEN, DMODEL, DMODEL);
    hipLaunchKernelGGL((gemm_wmma_kernel<0>), grid, block, 0, stream,
                       kin, wkt, b_k, (void*)Kb, BATCH * SLEN, DMODEL, DMODEL);
    hipLaunchKernelGGL((gemm_wmma_kernel<1>), grid, block, 0, stream,
                       vin, wvt, b_v, (void*)Vt, BATCH * SLEN, DMODEL, DMODEL);
  }

  // 3) attention: one WG per (b, h, 16-row q tile) = 4*16*128 = 8192 WGs
  {
    size_t smem = (size_t)16 * SLEN * 4   // fp32 scores
                + (size_t)16 * SLEN * 2   // bf16 probs
                + 2 * 16 * 8 * 4;         // reductions
    hipLaunchKernelGGL(attn_kernel, dim3(BATCH * NHEAD * (SLEN / 16)), dim3(128),
                       smem, stream, Qb, Kb, Vt, mask, sim_out, Xb);
  }

  // 4) output projection -> d_out fp32
  {
    dim3 grid(DMODEL / 64, (BATCH * SLEN) / 64);
    dim3 block(128);
    hipLaunchKernelGGL((gemm_wmma_kernel<2>), grid, block, 0, stream,
                       Xb, wot, b_o, (void*)out_proj, BATCH * SLEN, DMODEL, DMODEL);
  }
}
